// P_MLP_25374666784998
// MI455X (gfx1250) — matrix operand order repaired and checked by reading, not compiled, not run on hardware
//
#include <hip/hip_runtime.h>
#include <math.h>

// ---------------------------------------------------------------------------
// Problem constants (from the reference)
// ---------------------------------------------------------------------------
#define SQ      25
#define ENC     300
#define BATCH   256
#define H1      1024          // ARCHI[1]
#define N0      30000         // ARCHI[0] = 4*SQ*ENC
#define NCLS    10            // ARCHI[2]
#define TSTEPS  15
#define SCALE_F 0.16f

#define XLDA    304           // padded lda for 300-wide bf16 matrices (16B align)

typedef __attribute__((ext_vector_type(16))) __bf16 v16bf;
typedef __attribute__((ext_vector_type(8)))  __bf16 v8bf;
typedef __attribute__((ext_vector_type(8)))  float  v8f;

static __device__ __forceinline__ __bf16 toBf(float f) { return (__bf16)f; }

// ---------------------------------------------------------------------------
// CDNA5 async global->LDS copy support (ASYNCcnt-tracked, no VGPR roundtrip).
// Gated on builtin availability so the compile can never regress.
// Builtin signature (from clang diagnostics):
//   void __builtin_amdgcn_global_load_async_to_lds_b128(
//        int __attribute__((vector_size(16))) AS1* src,
//        int __attribute__((vector_size(16))) AS3* dst,
//        imm int offset, imm int cpol)
// ---------------------------------------------------------------------------
#if defined(__gfx1250__) && \
    __has_builtin(__builtin_amdgcn_global_load_async_to_lds_b128) && \
    __has_builtin(__builtin_amdgcn_s_wait_asynccnt)
#define USE_ASYNC_COPY 1
#else
#define USE_ASYNC_COPY 0
#endif

#if USE_ASYNC_COPY
typedef int v4i_vs __attribute__((vector_size(16)));
typedef __attribute__((address_space(1))) v4i_vs* g_b128_ptr;
typedef __attribute__((address_space(3))) v4i_vs* l_b128_ptr;
// Flat LDS addresses carry the LDS byte offset in their low 32 bits
// AS3 pointers are 32-bit.
static __device__ __forceinline__ l_b128_ptr lds_cast(void* p) {
    return (l_b128_ptr)(unsigned int)(unsigned long long)p;
}
static __device__ __forceinline__ g_b128_ptr gbl_cast(const void* p) {
    return (g_b128_ptr)(unsigned long long)p;
}
#endif

// ---------------------------------------------------------------------------
// Generic bf16 WMMA GEMM:  C[M,N] (+)= A[M,K] * Bt[N,K]^T
//   A   : bf16 row-major, lda (elements, multiple of 8)
//   Bt  : bf16 "B-transposed" row-major (row n holds column n of B), ldb mult 8
//   C   : fp32 row-major
//   Split-K via gridDim.z / kChunk with atomic accumulation.
// Block: 256 threads = 8 waves; block tile 128x128, BK=32;
// wave tile 32x64 = 2x4 v_wmma_f32_16x16x32_bf16 fragments.
// LDS is double-buffered; full tiles are staged with async global->LDS
// copies issued one K-step ahead (s_wait_asynccnt pipelining).
// ---------------------------------------------------------------------------
__global__ __launch_bounds__(256)
void gemm_bf16_wmma(const unsigned short* __restrict__ Araw, int lda,
                    const unsigned short* __restrict__ Btraw, int ldb,
                    float* __restrict__ C, int ldc,
                    const float* __restrict__ bias,
                    int M, int N, int K, int kChunk, int atomicAcc)
{
    const __bf16* __restrict__ A  = (const __bf16*)Araw;
    const __bf16* __restrict__ Bt = (const __bf16*)Btraw;

    __shared__ __align__(16) __bf16 As[2][128][40];   // 2 x (128 rows x 32 k)
    __shared__ __align__(16) __bf16 Bs[2][128][40];   // 2 x (128 cols x 32 k)

    const int tid  = threadIdx.x;
    const int lane = tid & 31;
    const int wid  = tid >> 5;
    const int wm   = (wid >> 1) * 32;     // wave row offset inside block tile
    const int wn   = (wid & 1) * 64;      // wave col offset inside block tile
    const int bm   = blockIdx.y * 128;
    const int bn   = blockIdx.x * 128;

    const int kbeg = blockIdx.z * kChunk;
    const int kend = (K < kbeg + kChunk) ? K : (kbeg + kChunk);
    if (kbeg >= kend) return;

    const int lrow  = tid >> 1;          // 0..127 : tile row loaded by this thread
    const int lhalf = (tid & 1) * 16;    // 0 or 16: which 16 k-elements

    // Stage one K-tile pair into buffer `buf`. Returns #async ops issued
    // (uniform across the wave: the async-vs-sync choice is tile-uniform).
    auto stage = [&](int buf, int k0) -> int {
        int issued = 0;
        const bool fullK = (k0 + 32 <= kend);
        // ---- A tile ----
#if USE_ASYNC_COPY
        if (fullK && (bm + 128 <= M)) {
            const __bf16* src = A + (size_t)(bm + lrow) * lda + (k0 + lhalf);
            char* dst = (char*)&As[buf][lrow][lhalf];
            __builtin_amdgcn_global_load_async_to_lds_b128(
                gbl_cast(src),     lds_cast(dst),      0, 0);
            __builtin_amdgcn_global_load_async_to_lds_b128(
                gbl_cast(src + 8), lds_cast(dst + 16), 0, 0);
            issued += 2;
        } else
#endif
        {
            const int gr = bm + lrow;
            const int gk = k0 + lhalf;
            if (gr < M && gk + 16 <= kend) {
                const uint4* src = (const uint4*)(A + (size_t)gr * lda + gk);
                *(uint4*)&As[buf][lrow][lhalf]     = src[0];
                *(uint4*)&As[buf][lrow][lhalf + 8] = src[1];
            } else {
                for (int j = 0; j < 16; ++j)
                    As[buf][lrow][lhalf + j] =
                        (gr < M && gk + j < kend) ? A[(size_t)gr * lda + gk + j]
                                                  : (__bf16)0.0f;
            }
        }
        // ---- Bt tile ----
#if USE_ASYNC_COPY
        if (fullK && (bn + 128 <= N)) {
            const __bf16* src = Bt + (size_t)(bn + lrow) * ldb + (k0 + lhalf);
            char* dst = (char*)&Bs[buf][lrow][lhalf];
            __builtin_amdgcn_global_load_async_to_lds_b128(
                gbl_cast(src),     lds_cast(dst),      0, 0);
            __builtin_amdgcn_global_load_async_to_lds_b128(
                gbl_cast(src + 8), lds_cast(dst + 16), 0, 0);
            issued += 2;
        } else
#endif
        {
            const int gn = bn + lrow;
            const int gk = k0 + lhalf;
            if (gn < N && gk + 16 <= kend) {
                const uint4* src = (const uint4*)(Bt + (size_t)gn * ldb + gk);
                *(uint4*)&Bs[buf][lrow][lhalf]     = src[0];
                *(uint4*)&Bs[buf][lrow][lhalf + 8] = src[1];
            } else {
                for (int j = 0; j < 16; ++j)
                    Bs[buf][lrow][lhalf + j] =
                        (gn < N && gk + j < kend) ? Bt[(size_t)gn * ldb + gk + j]
                                                  : (__bf16)0.0f;
            }
        }
        return issued;
    };

    v8f acc[2][4];
    #pragma unroll
    for (int mi = 0; mi < 2; ++mi)
        #pragma unroll
        for (int ni = 0; ni < 4; ++ni) acc[mi][ni] = (v8f){};

    stage(0, kbeg);                       // prologue: fill buffer 0

    int iter = 0;
    for (int k0 = kbeg; k0 < kend; k0 += 32, ++iter) {
        const int cur = iter & 1;
        int nNext = 0;
        if (k0 + 32 < kend) nNext = stage(cur ^ 1, k0 + 32);  // prefetch next
        (void)nNext;
#if USE_ASYNC_COPY
        // FIFO completion: wait until only the just-issued (next-buffer) ops
        // remain outstanding -> current buffer is resident in LDS.
        if (nNext >= 4)      __builtin_amdgcn_s_wait_asynccnt(4);
        else if (nNext >= 2) __builtin_amdgcn_s_wait_asynccnt(2);
        else                 __builtin_amdgcn_s_wait_asynccnt(0);
#endif
        __syncthreads();                  // publish buf[cur] to all waves

        // ---- fragments per documented 16-bit WMMA VGPR layouts ----
        const int aoff = (lane < 16) ? 0 : 8;   // A: K {0..7,16..23} | {8..15,24..31}
        const int koff = (lane < 16) ? 0 : 8;   // B: K {0..7,16..23} | {8..15,24..31}
        v16bf af[2], bf_[4];
        #pragma unroll
        for (int mi = 0; mi < 2; ++mi) {
            const int arow = wm + mi * 16 + (lane & 15);
            v8bf lo = *(const v8bf*)&As[cur][arow][aoff];
            v8bf hi = *(const v8bf*)&As[cur][arow][16 + aoff];
            af[mi] = __builtin_shufflevector(lo, hi, 0,1,2,3,4,5,6,7,
                                                      8,9,10,11,12,13,14,15);
        }
        #pragma unroll
        for (int ni = 0; ni < 4; ++ni) {
            const int brow = wn + ni * 16 + (lane & 15);
            v8bf lo = *(const v8bf*)&Bs[cur][brow][koff];
            v8bf hi = *(const v8bf*)&Bs[cur][brow][koff + 16];
            bf_[ni] = __builtin_shufflevector(lo, hi, 0,1,2,3,4,5,6,7,
                                                       8,9,10,11,12,13,14,15);
        }
        #pragma unroll
        for (int mi = 0; mi < 2; ++mi)
            #pragma unroll
            for (int ni = 0; ni < 4; ++ni)
                acc[mi][ni] = __builtin_amdgcn_wmma_f32_16x16x32_bf16(
                    false, af[mi], false, bf_[ni], (short)0, acc[mi][ni],
                    false, false);

        __syncthreads();                  // all reads of buf[cur] done
    }

    // ---- epilogue: C/D layout -> (M = vgpr + (lane>=16?8:0), N = lane&15) ----
    const int mbase = bm + wm + ((lane >= 16) ? 8 : 0);
    const int nbase = bn + wn + (lane & 15);
    #pragma unroll
    for (int mi = 0; mi < 2; ++mi) {
        #pragma unroll
        for (int ni = 0; ni < 4; ++ni) {
            const int n = nbase + ni * 16;
            if (n >= N) continue;
            #pragma unroll
            for (int i = 0; i < 8; ++i) {
                const int m = mbase + mi * 16 + i;
                if (m >= M) continue;
                float v = acc[mi][ni][i];
                if (atomicAcc) {
                    atomicAdd(&C[(size_t)m * ldc + n], v);
                } else {
                    if (bias) v += bias[n];
                    C[(size_t)m * ldc + n] = v;
                }
            }
        }
    }
}

// ---------------------------------------------------------------------------
// Fused attention: one block per (b, idx).
//   g  = P[idx][b] + gv[b, idx]           (25x300)
//   kv = xa or xb                          (25x300)
//   out = softmax(SCALE * g kv^T) kv  -> Nbig[b][idx], vb (bf16 copy)
// ---------------------------------------------------------------------------
__global__ __launch_bounds__(256)
void attn_kernel(const float* __restrict__ gv,    // [B][30000]
                 const float* __restrict__ P,     // [4][B*SQ][ENC]
                 const float* __restrict__ x,     // [B][2*SQ][ENC]
                 float* __restrict__ Nbig,        // [B][4][SQ][ENC]
                 unsigned short* __restrict__ vbr)// bf16 [B][30000]
{
    __bf16* __restrict__ vb = (__bf16*)vbr;
    __shared__ float g[SQ][ENC];
    __shared__ float kv[SQ][ENC];
    __shared__ float p[SQ][SQ];

    const int b   = blockIdx.x;
    const int idx = blockIdx.y;
    const int tid = threadIdx.x;
    const int kvsel = (idx == 0 || idx == 3) ? 1 : 0;   // 1 -> xb, 0 -> xa
    const float* Pi = P + (size_t)idx * (BATCH * SQ) * ENC;

    for (int t = tid; t < SQ * ENC; t += 256) {
        const int s = t / ENC, e = t % ENC;
        g[s][e]  = Pi[((size_t)b * SQ + s) * ENC + e]
                 + gv[(size_t)b * N0 + idx * (SQ * ENC) + t];
        kv[s][e] = x[((size_t)b * 2 * SQ + kvsel * SQ + s) * ENC + e];
    }
    __syncthreads();

    for (int t = tid; t < SQ * SQ; t += 256) {
        const int i = t / SQ, j = t % SQ;
        float s = 0.f;
        for (int e = 0; e < ENC; ++e) s += g[i][e] * kv[j][e];
        p[i][j] = SCALE_F * s;
    }
    __syncthreads();

    if (tid < SQ) {
        float mx = -3.0e38f;
        for (int j = 0; j < SQ; ++j) mx = fmaxf(mx, p[tid][j]);
        float sum = 0.f;
        for (int j = 0; j < SQ; ++j) { float ex = __expf(p[tid][j] - mx); p[tid][j] = ex; sum += ex; }
        const float inv = 1.f / sum;
        for (int j = 0; j < SQ; ++j) p[tid][j] *= inv;
    }
    __syncthreads();

    for (int t = tid; t < SQ * ENC; t += 256) {
        const int i = t / ENC, e = t % ENC;
        float o = 0.f;
        for (int j = 0; j < SQ; ++j) o += p[i][j] * kv[j][e];
        Nbig[(((size_t)b * 4 + idx) * SQ + i) * ENC + e] = o;
        vb[(size_t)b * N0 + idx * (SQ * ENC) + t] = toBf(o);
    }
}

// ---------------------------------------------------------------------------
// Small helper kernels
// ---------------------------------------------------------------------------
__global__ void g5_init_kernel(const float* __restrict__ n6,
                               const float* __restrict__ W5,
                               const float* __restrict__ b4,
                               float* __restrict__ g5)
{
    const int t = blockIdx.x * blockDim.x + threadIdx.x;
    if (t >= BATCH * H1) return;
    const int b = t / H1, n = t % H1;
    float v = b4[n];
    #pragma unroll
    for (int k = 0; k < NCLS; ++k) v += n6[b * NCLS + k] * W5[k * H1 + n];
    g5[t] = v;
}

__global__ void g6_tanh_kernel(const float* __restrict__ n5,
                               const float* __restrict__ W5,
                               const float* __restrict__ b5,
                               float* __restrict__ n6)
{
    const int t = blockIdx.x * blockDim.x + threadIdx.x;
    if (t >= BATCH * NCLS) return;
    const int b = t / NCLS, j = t % NCLS;
    float v = b5[j];
    for (int k = 0; k < H1; ++k) v += n5[b * H1 + k] * W5[j * H1 + k];
    n6[t] = tanhf(v);
}

__global__ void tanh5_kernel(const float* __restrict__ g5,
                             float* __restrict__ n5,
                             unsigned short* __restrict__ n5br)
{
    __bf16* n5b = (__bf16*)n5br;
    const int t = blockIdx.x * blockDim.x + threadIdx.x;
    if (t >= BATCH * H1) return;
    const float v = tanhf(g5[t]);
    n5[t]  = v;
    n5b[t] = toBf(v);
}

__global__ void cast_w4_kernel(const float* __restrict__ W4,
                               unsigned short* __restrict__ W4br,
                               unsigned short* __restrict__ W4Tbr)
{
    __bf16* W4b  = (__bf16*)W4br;
    __bf16* W4Tb = (__bf16*)W4Tbr;
    const int t = blockIdx.x * blockDim.x + threadIdx.x;
    if (t >= H1 * N0) return;
    const int r = t / N0, c = t % N0;
    const float v = W4[t];
    W4b[t]                    = toBf(v);   // [1024][30000]
    W4Tb[(size_t)c * H1 + r]  = toBf(v);   // [30000][1024]
}

__global__ void cast_w300_kernel(const float* __restrict__ W,
                                 unsigned short* __restrict__ Wbr)
{
    __bf16* Wb = (__bf16*)Wbr;
    const int t = blockIdx.x * blockDim.x + threadIdx.x;
    if (t >= ENC * ENC) return;
    const int r = t / ENC, c = t % ENC;
    Wb[r * XLDA + c] = toBf(W[t]);
}

__global__ void cast_x_kernel(const float* __restrict__ x,
                              unsigned short* __restrict__ Xabr,
                              unsigned short* __restrict__ Xbbr)
{
    __bf16* Xab = (__bf16*)Xabr;
    __bf16* Xbb = (__bf16*)Xbbr;
    const int t = blockIdx.x * blockDim.x + threadIdx.x;
    if (t >= BATCH * SQ * ENC) return;
    const int r = t / ENC, c = t % ENC;
    const int b = r / SQ, s = r % SQ;
    Xab[(size_t)r * XLDA + c] = toBf(x[((size_t)b * 2 * SQ + s) * ENC + c]);
    Xbb[(size_t)r * XLDA + c] = toBf(x[((size_t)b * 2 * SQ + SQ + s) * ENC + c]);
}

__global__ void zero_u32_kernel(unsigned int* __restrict__ p, long n)
{
    const long t = (long)blockIdx.x * blockDim.x + threadIdx.x;
    if (t < n) p[t] = 0u;
}

__global__ void finalize_kernel(const float* __restrict__ Nbig,
                                const float* __restrict__ n5,
                                const float* __restrict__ n6,
                                float* __restrict__ out)
{
    const long t  = (long)blockIdx.x * blockDim.x + threadIdx.x;
    const long NB = (long)BATCH * 4 * SQ * ENC;            // 7,680,000
    if (t < NB) {
        const int i = (int)(t / ((long)BATCH * SQ * ENC));
        const long r = t % ((long)BATCH * SQ * ENC);
        const int b = (int)(r / (SQ * ENC));
        const int q = (int)(r % (SQ * ENC));
        out[t] = Nbig[((long)b * 4 + i) * (SQ * ENC) + q];
    } else if (t < NB + (long)BATCH * H1) {
        out[t] = n5[t - NB];
    } else if (t < NB + (long)BATCH * H1 + (long)BATCH * NCLS) {
        out[t] = n6[t - NB - (long)BATCH * H1];
    }
}

// ---------------------------------------------------------------------------
// Host orchestration. Needs ~240 MB of workspace.
// ---------------------------------------------------------------------------
extern "C" void kernel_launch(void* const* d_in, const int* in_sizes, int n_in,
                              void* d_out, int out_size, void* d_ws, size_t ws_size,
                              hipStream_t stream)
{
    const float* x  = (const float*)d_in[0];
    // d_in[1]=y (unused), d_in[2..7]=n1..n6 initial zeros (we re-zero ourselves)
    const float* W0 = (const float*)d_in[8];
    const float* W1 = (const float*)d_in[9];
    const float* W2 = (const float*)d_in[10];
    const float* W3 = (const float*)d_in[11];
    const float* W4 = (const float*)d_in[12];
    const float* b4 = (const float*)d_in[13];
    const float* W5 = (const float*)d_in[14];
    const float* b5 = (const float*)d_in[15];
    // d_in[16] = T (device scalar; fixed to 15 by the reference setup)

    char* ws = (char*)d_ws;
    size_t off = 0;
    auto carve = [&](size_t bytes) -> void* {
        off = (off + 255) & ~(size_t)255;
        void* p = ws + off; off += bytes; return p;
    };

    unsigned short* W4b  = (unsigned short*)carve((size_t)H1 * N0 * 2);        // [1024][30000] bf16
    unsigned short* W4Tb = (unsigned short*)carve((size_t)N0 * H1 * 2);        // [30000][1024] bf16
    unsigned short* Wb[4];
    for (int i = 0; i < 4; ++i) Wb[i] = (unsigned short*)carve((size_t)ENC * XLDA * 2);
    unsigned short* Xab  = (unsigned short*)carve((size_t)BATCH * SQ * XLDA * 2);
    unsigned short* Xbb  = (unsigned short*)carve((size_t)BATCH * SQ * XLDA * 2);
    float* P    = (float*)carve((size_t)4 * BATCH * SQ * ENC * 4);             // [4][6400][300]
    float* gv   = (float*)carve((size_t)BATCH * N0 * 4);                       // [256][30000]
    float* g5   = (float*)carve((size_t)BATCH * H1 * 4);
    float* Nbig = (float*)carve((size_t)BATCH * 4 * SQ * ENC * 4);             // [256][4][25][300]
    unsigned short* vb  = (unsigned short*)carve((size_t)BATCH * N0 * 2);      // bf16 [256][30000]
    float* n5   = (float*)carve((size_t)BATCH * H1 * 4);
    unsigned short* n5b = (unsigned short*)carve((size_t)BATCH * H1 * 2);
    float* n6   = (float*)carve((size_t)BATCH * NCLS * 4);
    (void)ws_size; (void)in_sizes; (void)n_in; (void)out_size;

    const dim3 blk(256);
    auto g1d = [](long n) { return dim3((unsigned)((n + 255) / 256)); };

    // ---- zero-initialize recurrent state (reference carries start at 0) ----
    zero_u32_kernel<<<g1d((long)BATCH * 4 * SQ * ENC),     blk, 0, stream>>>((unsigned*)Nbig, (long)BATCH * 4 * SQ * ENC);
    zero_u32_kernel<<<g1d((long)BATCH * N0 / 2),           blk, 0, stream>>>((unsigned*)vb,   (long)BATCH * N0 / 2);
    zero_u32_kernel<<<g1d((long)BATCH * H1),               blk, 0, stream>>>((unsigned*)n5,   (long)BATCH * H1);
    zero_u32_kernel<<<g1d((long)BATCH * H1 / 2),           blk, 0, stream>>>((unsigned*)n5b,  (long)BATCH * H1 / 2);
    zero_u32_kernel<<<g1d((long)BATCH * NCLS),             blk, 0, stream>>>((unsigned*)n6,   (long)BATCH * NCLS);

    // ---- one-time precompute: bf16 weights/activations + projections ----
    cast_w4_kernel<<<g1d((long)H1 * N0), blk, 0, stream>>>(W4, W4b, W4Tb);
    cast_w300_kernel<<<g1d((long)ENC * ENC), blk, 0, stream>>>(W0, Wb[0]);
    cast_w300_kernel<<<g1d((long)ENC * ENC), blk, 0, stream>>>(W1, Wb[1]);
    cast_w300_kernel<<<g1d((long)ENC * ENC), blk, 0, stream>>>(W2, Wb[2]);
    cast_w300_kernel<<<g1d((long)ENC * ENC), blk, 0, stream>>>(W3, Wb[3]);
    cast_x_kernel<<<g1d((long)BATCH * SQ * ENC), blk, 0, stream>>>(x, Xab, Xbb);

    // P[i] = X{a|b} @ Wi^T : M=6400, N=300, K=300
    {
        const dim3 grid((300 + 127) / 128, (BATCH * SQ + 127) / 128, 1);
        const unsigned short* Asrc[4] = { Xab, Xbb, Xab, Xbb };
        for (int i = 0; i < 4; ++i) {
            gemm_bf16_wmma<<<grid, blk, 0, stream>>>(
                Asrc[i], XLDA, Wb[i], XLDA,
                P + (size_t)i * BATCH * SQ * ENC, ENC,
                nullptr, BATCH * SQ, ENC, ENC, ENC, 0);
        }
    }

    // ---- recurrence (single-buffered: all reads precede their overwrites) ----
    for (int t = 0; t < TSTEPS; ++t) {
        // gv = n5 @ W4            M=256, N=30000, K=1024
        {
            const dim3 grid((N0 + 127) / 128, (BATCH + 127) / 128, 1);
            gemm_bf16_wmma<<<grid, blk, 0, stream>>>(
                n5b, H1, W4Tb, H1, gv, N0, nullptr,
                BATCH, N0, H1, H1, 0);
        }
        // g5 = b4 + n6 @ W5   (cheap, reads OLD n6)
        g5_init_kernel<<<g1d((long)BATCH * H1), blk, 0, stream>>>(n6, W5, b4, g5);
        // g5 += v @ W4^T          M=256, N=1024, K=30000, split-K=30 (atomic)
        {
            const dim3 grid((H1 + 127) / 128, (BATCH + 127) / 128, 30);
            gemm_bf16_wmma<<<grid, blk, 0, stream>>>(
                vb, N0, W4b, N0, g5, H1, nullptr,
                BATCH, H1, N0, 1024, 1);
        }
        // n6 <- tanh(n5_old @ W5^T + b5)   (reads OLD n5; g5_init already used old n6)
        g6_tanh_kernel<<<g1d((long)BATCH * NCLS), blk, 0, stream>>>(n5, W5, b5, n6);
        // n1..n4 <- attention (reads gv; overwrites Nbig/vb AFTER gemm2 consumed vb)
        {
            const dim3 grid(BATCH, 4, 1);
            attn_kernel<<<grid, blk, 0, stream>>>(gv, P, x, Nbig, vb);
        }
        // n5 <- tanh(g5)  (after gemm1 & g6 consumed old n5)
        tanh5_kernel<<<g1d((long)BATCH * H1), blk, 0, stream>>>(g5, n5, n5b);
    }

    // ---- pack outputs: n1,n2,n3,n4,n5,n6 flat in return order ----
    const long total = (long)BATCH * 4 * SQ * ENC + (long)BATCH * H1 + (long)BATCH * NCLS;
    finalize_kernel<<<g1d(total), blk, 0, stream>>>(Nbig, n5, n6, (float*)d_out);
}
